// GCN_pignn_63204738728380
// MI455X (gfx1250) — matrix operand order repaired and checked by reading, not compiled, not run on hardware
//
#include <hip/hip_runtime.h>

typedef __attribute__((ext_vector_type(16))) __bf16 v16bf;
typedef __attribute__((ext_vector_type(8)))  float  v8f;
typedef __attribute__((ext_vector_type(4)))  float  vf4;
typedef __attribute__((ext_vector_type(4)))  __bf16 vbf4;

union FragAB { v16bf v; uint4 q[2]; };
union PackBF { vbf4 h[2]; uint4 q; };

__device__ __forceinline__ unsigned short f2bf(float f) {
  __bf16 h = (__bf16)f;                       // native v_cvt to bf16 (RNE)
  return __builtin_bit_cast(unsigned short, h);
}

#define BM 128
#define BN 64
#define BK 64
#define LP 72   // padded LDS row stride (elements) -> 144B, 16B-aligned, conflict-free

// Generic bf16-WMMA GEMM: C = A(f32,row-major) * B(bf16) [+bias][relu]
// BT=false: B is row-major [K][N] bf16, transposed while staging into LDS.
// BT=true : B is given as Z[N][K] bf16 (i.e. we compute A * Z^T), rows copied straight.
// EPI: 0 = f32 store; 1 = +bias -> bf16 store; 2 = relu -> f32 store; 3 = f32 + bf16 store.
template<int EPI, bool BT>
__global__ __launch_bounds__(256) void gemm_k(
    const float* __restrict__ A, const unsigned short* __restrict__ B,
    const float* __restrict__ bias,
    float* __restrict__ outF, unsigned short* __restrict__ outBf,
    int M, int N, int K, int lda, int ldb)
{
  __shared__ unsigned short lds_a[BM * LP];   // [m][k]   bf16
  __shared__ unsigned short lds_b[BN * LP];   // [n][k]   bf16 (transposed)

  const int tid  = threadIdx.x;
  const int n0   = blockIdx.x * BN;
  const int m0   = blockIdx.y * BM;
  const int w    = tid >> 5;        // wave 0..7
  const int wm   = (w >> 1) * 32;   // wave m offset: 0,32,64,96
  const int wn   = (w & 1) * 32;    // wave n offset: 0,32
  const int lane = tid & 31;
  const int l15  = lane & 15;
  const int kh   = lane >> 4;

  // staging coordinates
  const int arow  = tid >> 1;           // 0..127
  const int acol0 = (tid & 1) * 32;     // 0 or 32
  const int bk0   = tid >> 3;           // 0..31
  const int bn8   = (tid & 7) * 8;      // 0..56
  const int zn    = tid >> 2;           // 0..63
  const int zc    = (tid & 3) * 16;     // 0,16,32,48

  v8f acc[2][2] = {};

  for (int kt = 0; kt < K; kt += BK) {
    // ---- stage A tile: f32 global -> bf16 LDS (native packed converts) ----
    #pragma unroll
    for (int c = 0; c < 4; ++c) {
      const int col = acol0 + c * 8;
      const vf4* src = reinterpret_cast<const vf4*>(
          A + (size_t)(m0 + arow) * lda + kt + col);
      vf4 f0 = src[0], f1 = src[1];
      PackBF t;
      t.h[0] = __builtin_convertvector(f0, vbf4);
      t.h[1] = __builtin_convertvector(f1, vbf4);
      *reinterpret_cast<uint4*>(&lds_a[arow * LP + col]) = t.q;
    }
    // ---- stage B tile into LDS transposed [n][k] ----
    if (BT) {
      const uint4* src = reinterpret_cast<const uint4*>(
          B + (size_t)(n0 + zn) * ldb + kt + zc);
      *reinterpret_cast<uint4*>(&lds_b[zn * LP + zc])     = src[0];
      *reinterpret_cast<uint4*>(&lds_b[zn * LP + zc + 8]) = src[1];
    } else {
      #pragma unroll
      for (int it = 0; it < 2; ++it) {
        const int k = bk0 + it * 32;
        const uint4 src = *reinterpret_cast<const uint4*>(
            B + (size_t)(kt + k) * ldb + n0 + bn8);
        const unsigned short* e = reinterpret_cast<const unsigned short*>(&src);
        #pragma unroll
        for (int i = 0; i < 8; ++i) lds_b[(bn8 + i) * LP + k] = e[i];
      }
    }
    if (kt + BK < K) {  // global_prefetch_b8 for the next A tile
      __builtin_prefetch(A + (size_t)(m0 + arow) * lda + kt + BK + acol0, 0, 0);
    }
    __syncthreads();

    // ---- 2 K-substeps x 4 WMMA = 8 WMMA per barrier ----
    #pragma unroll
    for (int ks = 0; ks < 2; ++ks) {
      const int kb = ks * 32;
      FragAB fa[2], fb[2];
      #pragma unroll
      for (int ai = 0; ai < 2; ++ai) {
        const unsigned short* ap = &lds_a[(wm + ai * 16 + l15) * LP + kb];
        fa[ai].q[0] = *reinterpret_cast<const uint4*>(ap + kh * 8);
        fa[ai].q[1] = *reinterpret_cast<const uint4*>(ap + 16 + kh * 8);
      }
      #pragma unroll
      for (int bi = 0; bi < 2; ++bi) {
        const unsigned short* bp = &lds_b[(wn + bi * 16 + l15) * LP + kb + kh * 8];
        fb[bi].q[0] = *reinterpret_cast<const uint4*>(bp);
        fb[bi].q[1] = *reinterpret_cast<const uint4*>(bp + 16);
      }
      #pragma unroll
      for (int ai = 0; ai < 2; ++ai)
        #pragma unroll
        for (int bi = 0; bi < 2; ++bi)
          acc[ai][bi] = __builtin_amdgcn_wmma_f32_16x16x32_bf16(
              false, fa[ai].v, false, fb[bi].v, (short)0, acc[ai][bi], false, false);
    }
    __syncthreads();
  }

  // ---- epilogue: C/D layout M = v + 8*kh, N = l15 ----
  #pragma unroll
  for (int ai = 0; ai < 2; ++ai) {
    #pragma unroll
    for (int v = 0; v < 8; ++v) {
      const int row = m0 + wm + ai * 16 + 8 * kh + v;
      #pragma unroll
      for (int bi = 0; bi < 2; ++bi) {
        const int col = n0 + wn + bi * 16 + l15;
        float val = acc[ai][bi][v];
        const size_t off = (size_t)row * N + col;
        if (EPI == 1)      { val += bias[col]; outBf[off] = f2bf(val); }
        else if (EPI == 2) { outF[off] = fmaxf(val, 0.0f); }
        else if (EPI == 3) { outF[off] = val; outBf[off] = f2bf(val); }
        else               { outF[off] = val; }
      }
    }
  }
}

__global__ void cvt_bf16_k(const float* __restrict__ in,
                           unsigned short* __restrict__ out, int n) {
  int i = blockIdx.x * blockDim.x + threadIdx.x;
  if (i < n) out[i] = f2bf(in[i]);
}

// one wave per row of z[rows][64]; lane covers cols {lane, lane+32}
__global__ __launch_bounds__(256) void logsoftmax_k(const float* __restrict__ z,
                                                    float* __restrict__ out, int rows) {
  int row  = (blockIdx.x * 256 + threadIdx.x) >> 5;
  int lane = threadIdx.x & 31;
  if (row >= rows) return;
  float a = z[row * 64 + lane];
  float b = z[row * 64 + 32 + lane];
  float mx = fmaxf(a, b);
  #pragma unroll
  for (int o = 16; o > 0; o >>= 1) mx = fmaxf(mx, __shfl_xor(mx, o, 32));
  float s = __expf(a - mx) + __expf(b - mx);
  #pragma unroll
  for (int o = 16; o > 0; o >>= 1) s += __shfl_xor(s, o, 32);
  float lse = mx + __logf(s);
  out[row * 64 + lane]      = a - lse;
  out[row * 64 + 32 + lane] = b - lse;
}

extern "C" void kernel_launch(void* const* d_in, const int* in_sizes, int n_in,
                              void* d_out, int out_size, void* d_ws, size_t ws_size,
                              hipStream_t stream) {
  const float* x   = (const float*)d_in[0];
  const float* adj = (const float*)d_in[1];
  const float* W1  = (const float*)d_in[2];
  const float* b1  = (const float*)d_in[3];
  const float* W2  = (const float*)d_in[4];
  const float* b2  = (const float*)d_in[5];
  float* out = (float*)d_out;

  constexpr int N = 8192, FIN = 512, FHID = 256, FOUT = 64;

  char* ws = (char*)d_ws;
  size_t off = 0;
  auto alloc = [&](size_t bytes) {
    char* p = ws + off;
    off = (off + bytes + 255) & ~(size_t)255;
    return p;
  };
  unsigned short* W1bf = (unsigned short*)alloc((size_t)FIN * FHID * 2);
  unsigned short* W2bf = (unsigned short*)alloc((size_t)FHID * FOUT * 2);
  unsigned short* Sbf  = (unsigned short*)alloc((size_t)N * FHID * 2);  // support bf16
  float*          Hf   = (float*)         alloc((size_t)N * FHID * 4); // h f32
  unsigned short* Tbf  = (unsigned short*)alloc((size_t)N * FOUT * 2); // h@W2+b2 bf16
  float*          Zf   = (float*)         alloc((size_t)N * FOUT * 4); // z f32
  unsigned short* Zbf  = (unsigned short*)alloc((size_t)N * FOUT * 2); // z bf16

  cvt_bf16_k<<<(FIN * FHID + 255) / 256, 256, 0, stream>>>(W1, W1bf, FIN * FHID);
  cvt_bf16_k<<<(FHID * FOUT + 255) / 256, 256, 0, stream>>>(W2, W2bf, FHID * FOUT);

  dim3 blk(256);
  // support_bf = bf16(x @ W1 + b1)          [8192 x 256]
  gemm_k<1, false><<<dim3(FHID / BN, N / BM), blk, 0, stream>>>(
      x, W1bf, b1, nullptr, Sbf, N, FHID, FIN, FIN, FHID);
  // h = relu(adj @ support)                 [8192 x 256] f32
  gemm_k<2, false><<<dim3(FHID / BN, N / BM), blk, 0, stream>>>(
      adj, Sbf, nullptr, Hf, nullptr, N, FHID, N, N, FHID);
  // t_bf = bf16(h @ W2 + b2)                [8192 x 64]
  gemm_k<1, false><<<dim3(FOUT / BN, N / BM), blk, 0, stream>>>(
      Hf, W2bf, b2, nullptr, Tbf, N, FOUT, FHID, FHID, FOUT);
  // z = adj @ t  (f32 + bf16 copies)        [8192 x 64]
  gemm_k<3, false><<<dim3(FOUT / BN, N / BM), blk, 0, stream>>>(
      adj, Tbf, nullptr, Zf, Zbf, N, FOUT, N, N, FOUT);
  // x_product = z @ z^T                     [8192 x 8192] -> d_out tail
  gemm_k<0, true><<<dim3(N / BN, N / BM), blk, 0, stream>>>(
      Zf, Zbf, nullptr, out + (size_t)N * FOUT, nullptr, N, N, FOUT, FOUT, FOUT);
  // log_softmax(z) -> d_out head            [8192 x 64]
  logsoftmax_k<<<(N * 32) / 256, 256, 0, stream>>>(Zf, out, N);
}
